// GraphSAGEFraudDetector_20323785244835
// MI455X (gfx1250) — matrix operand order repaired and checked by reading, not compiled, not run on hardware
//
#include <hip/hip_runtime.h>
#include <math.h>

#define NNODES 100000
#define NEDGES 1600000
#define FDIM   64
#define NTILES (NNODES / 16)   // 6250, exact
#define BN_EPS 1e-5f

typedef __attribute__((ext_vector_type(16))) __bf16 v16bf;
typedef __attribute__((ext_vector_type(8)))  float  v8f;

// ---------------------------------------------------------------------------
// Degree count (float atomics so we can reuse the buffer as 1/deg)
// ---------------------------------------------------------------------------
__global__ void deg_kernel(const int* __restrict__ dst, float* __restrict__ cnt, int nE) {
  int e = blockIdx.x * blockDim.x + threadIdx.x;
  if (e < nE) atomicAdd(&cnt[dst[e]], 1.0f);
}

__global__ void inv_kernel(float* __restrict__ cnt, int n) {
  int i = blockIdx.x * blockDim.x + threadIdx.x;
  if (i < n) cnt[i] = 1.0f / fmaxf(cnt[i], 1.0f);
}

// ---------------------------------------------------------------------------
// Edge scatter-add: one thread per (edge, float4 chunk).
// 16 consecutive lanes share an edge -> coalesced 256B gather of h[src],
// row-coalesced global_atomic_add_f32 into aggr[dst].
// ---------------------------------------------------------------------------
__global__ void scatter_kernel(const float* __restrict__ h, const int* __restrict__ src,
                               const int* __restrict__ dst, float* __restrict__ aggr, int nE) {
  long idx   = (long)blockIdx.x * blockDim.x + threadIdx.x;
  long total = (long)nE * 16;
  if (idx >= total) return;
  int e = (int)(idx >> 4);
  int f = (int)(idx & 15) * 4;
  int s = src[e], d = dst[e];
  const float4 val = *(const float4*)(h + (long)s * FDIM + f);
  float* out = aggr + (long)d * FDIM + f;
  atomicAdd(out + 0, val.x);
  atomicAdd(out + 1, val.y);
  atomicAdd(out + 2, val.z);
  atomicAdd(out + 3, val.w);
}

// ---------------------------------------------------------------------------
// bf16 WMMA fragment loaders (wave32 layouts per CDNA5 ISA 7.12.2)
// A (16x32, MxK): lane L holds row M=L&15; K = kc + (lane>=16 ? 8:0) + {0..7}
//                 in elems 0..7, and +16 more in elems 8..15.
// B (32x16, KxN): lane L holds col N=L&15; K same as A (kc + (lane>=16 ? 8:0) + {0..7}, +16 in elems 8..15).
// ---------------------------------------------------------------------------
__device__ inline v16bf load_a_frag(const float* __restrict__ row, int kc, int hi, float scale) {
  v16bf a;
  const float* p0 = row + kc + hi * 8;
  const float* p1 = p0 + 16;
#pragma unroll
  for (int j = 0; j < 8; ++j) {
    a[j]     = (__bf16)(p0[j] * scale);
    a[j + 8] = (__bf16)(p1[j] * scale);
  }
  return a;
}

__device__ inline v16bf load_b_frag(const float* __restrict__ w, int ob, int kc, int lane) {
  int o  = ob + (lane & 15);
  int hi = lane >> 4;
  const float* p = w + (long)o * FDIM + kc + hi * 8;
  v16bf b;
#pragma unroll
  for (int j = 0; j < 8; ++j) { b[j] = (__bf16)p[j]; b[j + 8] = (__bf16)p[j + 16]; }
  return b;
}

// ---------------------------------------------------------------------------
// Fused SAGEConv GEMM: out = act( BN( aggr*inv @ Wl^T + b + h @ Wr^T ) )
// One wave32 per 16-node tile; 16 x v_wmma_f32_16x16x32_bf16 per tile.
// Wave-uniform tile guard keeps EXEC all-1s (WMMA requirement).
// ---------------------------------------------------------------------------
__global__ void sage_gemm_kernel(const float* __restrict__ hin, const float* __restrict__ aggr,
                                 const float* __restrict__ inv,
                                 const float* __restrict__ wl, const float* __restrict__ bl,
                                 const float* __restrict__ wr,
                                 const float* __restrict__ g,  const float* __restrict__ be,
                                 const float* __restrict__ mm, const float* __restrict__ vv,
                                 float* __restrict__ hout, int n_tiles, int act) {
  int lane = threadIdx.x & 31;
  int wave = blockIdx.x * (blockDim.x >> 5) + (threadIdx.x >> 5);
  if (wave >= n_tiles) return;          // uniform across the wave

  int base = wave * 16;
  int mrow = lane & 15;
  int hi   = lane >> 4;
  int node = base + mrow;

  float scale = inv[node];
  const float* arow = aggr + (long)node * FDIM;
  const float* hrow = hin  + (long)node * FDIM;

  // A fragments: scaled aggregate and root features, K = [0,32) and [32,64)
  v16bf a_ag[2], a_h[2];
#pragma unroll
  for (int kc = 0; kc < 2; ++kc) {
    a_ag[kc] = load_a_frag(arow, kc * 32, hi, scale);
    a_h[kc]  = load_a_frag(hrow, kc * 32, hi, 1.0f);
  }

#pragma unroll
  for (int t = 0; t < 4; ++t) {         // 4 output-column tiles of 16
    int ob = t * 16;
    v8f c = {};
#pragma unroll
    for (int kc = 0; kc < 2; ++kc) {
      v16bf bL = load_b_frag(wl, ob, kc * 32, lane);
      c = __builtin_amdgcn_wmma_f32_16x16x32_bf16(false, a_ag[kc], false, bL,
                                                  (short)0, c, false, false);
    }
#pragma unroll
    for (int kc = 0; kc < 2; ++kc) {
      v16bf bR = load_b_frag(wr, ob, kc * 32, lane);
      c = __builtin_amdgcn_wmma_f32_16x16x32_bf16(false, a_h[kc], false, bR,
                                                  (short)0, c, false, false);
    }

    // epilogue: bias (+ folded BN scale/shift + ReLU for layers 0,1)
    int   o    = ob + mrow;
    float bias = bl[o];
    float sc = 1.0f, sh = 0.0f;
    if (act) {
      float rs = rsqrtf(vv[o] + BN_EPS);
      sc = g[o] * rs;
      sh = be[o] - mm[o] * sc;
    }
#pragma unroll
    for (int r = 0; r < 8; ++r) {       // C layout: VGPR r -> row r + 8*hi
      float tval = c[r] + bias;
      if (act) tval = fmaxf(tval * sc + sh, 0.0f);
      hout[(long)(base + r + 8 * hi) * FDIM + o] = tval;
    }
  }
}

// ---------------------------------------------------------------------------
// Column sums of [N,64] for the global mean pool
// ---------------------------------------------------------------------------
__global__ void colsum_kernel(const float* __restrict__ h, float* __restrict__ colsum, int n) {
  int f      = threadIdx.x & 63;
  int part   = blockIdx.x * (blockDim.x >> 6) + (threadIdx.x >> 6);
  int stride = gridDim.x * (blockDim.x >> 6);
  float acc = 0.0f;
  for (int r = part; r < n; r += stride) acc += h[(long)r * FDIM + f];
  atomicAdd(&colsum[f], acc);
}

// ---------------------------------------------------------------------------
// Head: mean -> Linear(64,64)+ReLU -> Linear(64,1) -> sigmoid
// ---------------------------------------------------------------------------
__global__ void head_kernel(const float* __restrict__ colsum,
                            const float* __restrict__ cw1, const float* __restrict__ cb1,
                            const float* __restrict__ cw2, const float* __restrict__ cb2,
                            float* __restrict__ out) {
  __shared__ float mean[64];
  __shared__ float hid[64];
  int j = threadIdx.x;
  mean[j] = colsum[j] * (1.0f / (float)NNODES);
  __syncthreads();
  float acc = cb1[j];
#pragma unroll
  for (int k = 0; k < 64; ++k) acc += mean[k] * cw1[j * 64 + k];
  hid[j] = fmaxf(acc, 0.0f);
  __syncthreads();
  if (j == 0) {
    float o = cb2[0];
    for (int k = 0; k < 64; ++k) o += hid[k] * cw2[k];
    out[0] = 1.0f / (1.0f + expf(-o));
  }
}

// ---------------------------------------------------------------------------
extern "C" void kernel_launch(void* const* d_in, const int* in_sizes, int n_in,
                              void* d_out, int out_size, void* d_ws, size_t ws_size,
                              hipStream_t stream) {
  (void)in_sizes; (void)n_in; (void)out_size; (void)ws_size;

  const float* x   = (const float*)d_in[0];
  const int*   ei  = (const int*)d_in[1];
  const int*   src = ei;
  const int*   dst = ei + NEDGES;
  const float* wl0 = (const float*)d_in[2];
  const float* bl0 = (const float*)d_in[3];
  const float* wr0 = (const float*)d_in[4];
  const float* wl1 = (const float*)d_in[5];
  const float* bl1 = (const float*)d_in[6];
  const float* wr1 = (const float*)d_in[7];
  const float* wl2 = (const float*)d_in[8];
  const float* bl2 = (const float*)d_in[9];
  const float* wr2 = (const float*)d_in[10];
  const float* g0  = (const float*)d_in[11];
  const float* be0 = (const float*)d_in[12];
  const float* m0  = (const float*)d_in[13];
  const float* v0  = (const float*)d_in[14];
  const float* g1  = (const float*)d_in[15];
  const float* be1 = (const float*)d_in[16];
  const float* m1  = (const float*)d_in[17];
  const float* v1  = (const float*)d_in[18];
  const float* cw1 = (const float*)d_in[19];
  const float* cb1 = (const float*)d_in[20];
  const float* cw2 = (const float*)d_in[21];
  const float* cb2 = (const float*)d_in[22];

  // workspace layout
  const size_t SZH = (size_t)NNODES * FDIM * sizeof(float);   // 25.6 MB
  char* ws = (char*)d_ws;
  size_t off = 0;
  float* inv    = (float*)(ws + off); off += ((size_t)NNODES * sizeof(float) + 255) & ~(size_t)255;
  float* aggr   = (float*)(ws + off); off += SZH;
  float* hA     = (float*)(ws + off); off += SZH;
  float* hB     = (float*)(ws + off); off += SZH;
  float* colsum = (float*)(ws + off); off += 256;

  const int TPB        = 256;
  const int deg_grid   = (NEDGES + TPB - 1) / TPB;
  const int inv_grid   = (NNODES + TPB - 1) / TPB;
  const int scat_grid  = (int)(((long)NEDGES * 16 + TPB - 1) / TPB);
  const int gemm_grid  = (NTILES + (TPB / 32) - 1) / (TPB / 32);

  // --- degrees (shared by all layers) ---
  hipMemsetAsync(inv, 0, (size_t)NNODES * sizeof(float), stream);
  deg_kernel<<<deg_grid, TPB, 0, stream>>>(dst, inv, NEDGES);
  inv_kernel<<<inv_grid, TPB, 0, stream>>>(inv, NNODES);

  // --- layer 0: x -> hA (BN0 + ReLU) ---
  hipMemsetAsync(aggr, 0, SZH, stream);
  scatter_kernel<<<scat_grid, TPB, 0, stream>>>(x, src, dst, aggr, NEDGES);
  sage_gemm_kernel<<<gemm_grid, TPB, 0, stream>>>(x, aggr, inv, wl0, bl0, wr0,
                                                  g0, be0, m0, v0, hA, NTILES, 1);

  // --- layer 1: hA -> hB (BN1 + ReLU) ---
  hipMemsetAsync(aggr, 0, SZH, stream);
  scatter_kernel<<<scat_grid, TPB, 0, stream>>>(hA, src, dst, aggr, NEDGES);
  sage_gemm_kernel<<<gemm_grid, TPB, 0, stream>>>(hA, aggr, inv, wl1, bl1, wr1,
                                                  g1, be1, m1, v1, hB, NTILES, 1);

  // --- layer 2: hB -> hA (no BN / ReLU) ---
  hipMemsetAsync(aggr, 0, SZH, stream);
  scatter_kernel<<<scat_grid, TPB, 0, stream>>>(hB, src, dst, aggr, NEDGES);
  sage_gemm_kernel<<<gemm_grid, TPB, 0, stream>>>(hB, aggr, inv, wl2, bl2, wr2,
                                                  g0, be0, m0, v0, hA, NTILES, 0);

  // --- mean pool + MLP head ---
  hipMemsetAsync(colsum, 0, 64 * sizeof(float), stream);
  colsum_kernel<<<512, TPB, 0, stream>>>(hA, colsum, NNODES);
  head_kernel<<<1, 64, 0, stream>>>(colsum, cw1, cb1, cw2, cb2, (float*)d_out);
}
